// MeshInterpolateModuleVoGE_3307124818190
// MI455X (gfx1250) — matrix operand order repaired and checked by reading, not compiled, not run on hardware
//
#include <hip/hip_runtime.h>
#include <hip/hip_bf16.h>

typedef __attribute__((ext_vector_type(16))) _Float16 v16h;
typedef __attribute__((ext_vector_type(8)))  float    v8f;

#define NB   2
#define NV   1024
#define NF   2048
#define NC   128
#define NHW  128
#define FOCAL_C 2.0f

// ---------------------------------------------------------------------------
// 1) zero sigma accumulators
// ---------------------------------------------------------------------------
__global__ void k_zero(float* ssum, float* cnt) {
  int i = blockIdx.x * blockDim.x + threadIdx.x;
  if (i < NV) { ssum[i] = 0.0f; cnt[i] = 0.0f; }
}

// ---------------------------------------------------------------------------
// 2) edge-length segment sums (edge length is invariant under the [0,2,1]
//    coordinate permutation, so use vertices directly)
// ---------------------------------------------------------------------------
__global__ void k_edges(const float* __restrict__ verts,
                        const int*   __restrict__ faces,
                        float* ssum, float* cnt) {
  int f = blockIdx.x * blockDim.x + threadIdx.x;
  if (f >= NF) return;
  int i0 = faces[f * 3 + 0], i1 = faces[f * 3 + 1], i2 = faces[f * 3 + 2];
  float a0 = verts[i0*3+0], a1 = verts[i0*3+1], a2 = verts[i0*3+2];
  float b0 = verts[i1*3+0], b1 = verts[i1*3+1], b2 = verts[i1*3+2];
  float c0 = verts[i2*3+0], c1 = verts[i2*3+1], c2 = verts[i2*3+2];
  float e01 = sqrtf((a0-b0)*(a0-b0) + (a1-b1)*(a1-b1) + (a2-b2)*(a2-b2));
  float e12 = sqrtf((b0-c0)*(b0-c0) + (b1-c1)*(b1-c1) + (b2-c2)*(b2-c2));
  float e20 = sqrtf((c0-a0)*(c0-a0) + (c1-a1)*(c1-a1) + (c2-a2)*(c2-a2));
  atomicAdd(&ssum[i0], e01 + e20); atomicAdd(&cnt[i0], 2.0f);
  atomicAdd(&ssum[i1], e01 + e12); atomicAdd(&cnt[i1], 2.0f);
  atomicAdd(&ssum[i2], e12 + e20); atomicAdd(&cnt[i2], 2.0f);
}

// ---------------------------------------------------------------------------
// 3) camera + projection + sigma -> px, py, s2d, z  (unsorted, [b][n])
// ---------------------------------------------------------------------------
__global__ void k_project(const float* __restrict__ campos,
                          const float* __restrict__ theta,
                          const float* __restrict__ vertices,
                          const float* __restrict__ ssum,
                          const float* __restrict__ cnt,
                          float* px, float* py, float* s2d, float* zz) {
  int t = blockIdx.x * blockDim.x + threadIdx.x;
  if (t >= NB * NV) return;
  int b = t / NV, n = t % NV;

  float cx = campos[b*3+0], cy = campos[b*3+1], cz = campos[b*3+2];
  // z axis = normalize(at - campos), at = 0
  float zl = sqrtf(cx*cx + cy*cy + cz*cz) + 1e-8f;
  float Z0 = -cx / zl, Z1 = -cy / zl, Z2 = -cz / zl;
  // x axis = normalize(cross(up, z)), up = (0,1,0) -> (z2, 0, -z0)
  float xl = sqrtf(Z2*Z2 + Z0*Z0) + 1e-8f;
  float X0 = Z2 / xl, X1 = 0.0f, X2 = -Z0 / xl;
  // y axis = cross(z, x)
  float Y0 = Z1*X2 - Z2*X1, Y1 = Z2*X0 - Z0*X2, Y2 = Z0*X1 - Z1*X0;

  float cth = cosf(theta[b]), sth = sinf(theta[b]);
  // R0 = [x y z] (columns); R = R0 * Rz
  float Xv[3] = {X0, X1, X2}, Yv[3] = {Y0, Y1, Y2}, Zv[3] = {Z0, Z1, Z2};
  float CP[3] = {cx, cy, cz};
  float R[3][3];
  #pragma unroll
  for (int i = 0; i < 3; ++i) {
    R[i][0] =  Xv[i]*cth + Yv[i]*sth;
    R[i][1] = -Xv[i]*sth + Yv[i]*cth;
    R[i][2] =  Zv[i];
  }
  float T[3];
  #pragma unroll
  for (int i = 0; i < 3; ++i)
    T[i] = -(R[0][i]*CP[0] + R[1][i]*CP[1] + R[2][i]*CP[2]);

  // verts = vertices[:, [0,2,1]]
  float v0 = vertices[n*3+0], v1 = vertices[n*3+2], v2 = vertices[n*3+1];
  float vc[3];
  #pragma unroll
  for (int j = 0; j < 3; ++j)
    vc[j] = v0*R[0][j] + v1*R[1][j] + v2*R[2][j] + T[j];

  float zv = vc[2];
  float zc = fmaxf(zv, 1e-4f);
  float me = ssum[n] / fmaxf(cnt[n], 1.0f);
  float rad = 0.5f * me;
  float sig = rad * rad + 1e-6f;
  float fz = FOCAL_C / zc;
  px[t]  = FOCAL_C * vc[0] / zc;
  py[t]  = FOCAL_C * vc[1] / zc;
  s2d[t] = sig * fz * fz;
  zz[t]  = zv;
}

// ---------------------------------------------------------------------------
// 4) per-batch bitonic sort of (z, idx) in LDS; emit packed float4 params
//    par4[b][i] = (px, py, s2d, z) in depth order, plus the sorted index.
// ---------------------------------------------------------------------------
__global__ __launch_bounds__(NV) void k_sort(const float* __restrict__ px,
                                             const float* __restrict__ py,
                                             const float* __restrict__ s2d,
                                             const float* __restrict__ zz,
                                             float4* __restrict__ par4,
                                             int* __restrict__ sidx) {
  __shared__ float key[NV];
  __shared__ int   val[NV];
  int b = blockIdx.x, tid = threadIdx.x;
  key[tid] = zz[b * NV + tid];
  val[tid] = tid;
  __syncthreads();
  for (int k = 2; k <= NV; k <<= 1) {
    for (int j = k >> 1; j > 0; j >>= 1) {
      int ixj = tid ^ j;
      if (ixj > tid) {
        bool up = ((tid & k) == 0);
        float a = key[tid], c = key[ixj];
        if ((a > c) == up) {
          key[tid] = c; key[ixj] = a;
          int tv = val[tid]; val[tid] = val[ixj]; val[ixj] = tv;
        }
      }
      __syncthreads();
    }
  }
  int idx = val[tid];
  par4[b*NV + tid] = make_float4(px[b*NV + idx], py[b*NV + idx],
                                 s2d[b*NV + idx], key[tid]);
  sidx[b*NV + tid] = idx;
}

// ---------------------------------------------------------------------------
// 5) depth-sorted, transposed f16 feature matrix: featT[b][c][i]
//    -> WMMA B-fragments become contiguous 32B loads (L2-resident, 512 KB)
// ---------------------------------------------------------------------------
__global__ void k_featT(const float* __restrict__ memory,
                        const int* __restrict__ sidx,
                        _Float16* __restrict__ featT) {
  int t = blockIdx.x * blockDim.x + threadIdx.x;  // t = ((b*NC)+c)*NV + i
  if (t >= NB * NC * NV) return;
  int i = t % NV;
  int c = (t / NV) % NC;
  int b = t / (NV * NC);
  int n = sidx[b * NV + i];
  featT[t] = (_Float16)memory[n * NC + c];
}

// ---------------------------------------------------------------------------
// 6) fused splat + transmittance + WMMA feature GEMM + normalize
//    grid = (128 rows, B); block = 256 (8 waves). Wave w owns pixels
//    [w*16, w*16+16) of its row and all 128 channels (8 x 16x16 C tiles).
//    The lane pair (l, l+16) shares pixel l&15; each half computes 16 of the
//    32 K-vertices per chunk (no redundancy); the transmittance prefix is
//    stitched with one shfl_xor; A uses each half's own K (k = 16*hi + j).
// ---------------------------------------------------------------------------
__global__ __launch_bounds__(256) void k_main(const float4* __restrict__ par4,
                                              const _Float16* __restrict__ featT,
                                              float* __restrict__ out) {
  const int h    = blockIdx.x;
  const int b    = blockIdx.y;
  const int wave = threadIdx.x >> 5;
  const int lane = threadIdx.x & 31;
  const int m    = lane & 15;       // pixel within wave tile / D-matrix N idx
  const int hi   = lane >> 4;       // lane half: K sub-block owner

  const float step = 2.0f / (float)(NHW - 1);
  const float gx = -1.0f + step * (float)(wave * 16 + m);
  const float gy = -1.0f + step * (float)h;

  const float4* par = par4 + (size_t)b * NV;
  const _Float16* frow = featT + (size_t)b * NC * NV;

  float S = 0.0f;       // cumsum(w) over all processed vertices (this pixel)
  float normH = 0.0f;   // this half's share of sum(wf)
  v8f acc[8] = {};

  for (int kc = 0; kc < NV; kc += 32) {
    const int kbase = kc + (hi << 4);   // this half's 16 vertices

    // --- pass 1: raw gaussian weights for this half ---
    float wl[16];
    float halfsum = 0.0f;
    #pragma unroll
    for (int j = 0; j < 16; ++j) {
      float4 P = par[kbase + j];        // (px, py, s2d, z): b128 load
      float dx = P.x - gx, dy = P.y - gy;
      float d2 = dx * dx + dy * dy;
      float w = __expf(-d2 / (2.0f * P.z));
      w = (P.w > 1e-4f) ? w : 0.0f;
      wl[j] = w;
      halfsum += w;
    }

    // --- stitch prefix across the lane pair ---
    float other = __shfl_xor(halfsum, 16, 32);
    float base  = S + (hi ? other : 0.0f);  // exclusive prefix at local j=0
    S += halfsum + other;

    // --- pass 2: transmittance-weighted wf (in place) ---
    float run = 0.0f;
    #pragma unroll
    for (int j = 0; j < 16; ++j) {
      float w  = wl[j];
      float wf = w * __expf(-(base + run));
      run += w;
      normH += wf;
      wl[j] = wf;
    }

    // --- exchange 8 values (rx unused; A keeps each half's own K):
    //     lo lanes hold K {0..15}; hi lanes hold K {16..31} ---
    float rx[8];
    #pragma unroll
    for (int e = 0; e < 8; ++e) {
      float send = hi ? wl[e] : wl[8 + e];  // lo sends K=8+e, hi sends K=16+e
      rx[e] = __shfl_xor(send, 16, 32);
    }

    v16h A;
    #pragma unroll
    for (int e = 0; e < 8; ++e) {
      A[e]     = (_Float16)wl[e];      // K = 16*hi + e
      A[e + 8] = (_Float16)wl[8 + e];  // K = 16*hi + 8 + e
    }

    // --- B fragments (32x16 f16): lane l -> column l&15,
    //     K rows kc + (lane>=16 ? 16 : 0) + 0..15 -> 32B contiguous load ---
    #pragma unroll
    for (int t = 0; t < 8; ++t) {
      int c = t * 16 + m;
      v16h Bf = *(const v16h*)(frow + (size_t)c * NV + kbase);
      acc[t] = __builtin_amdgcn_wmma_f32_16x16x32_f16(
          /*neg_a=*/false, A, /*neg_b=*/false, Bf,
          /*c_mod=*/(short)0, acc[t], /*reuse_a=*/false, /*reuse_b=*/false);
    }
  }

  // total normalizer per pixel = both halves
  float normAcc = normH + __shfl_xor(normH, 16, 32);

  // D layout: VGPR r, lane l -> M = r + (l>=16?8:0), N = l&15.
  // norm for pixel p lives in lanes p and p+16.
  float invr[8];
  #pragma unroll
  for (int r = 0; r < 8; ++r) {
    float nr = __shfl(normAcc, (lane < 16) ? r : (r + 24), 32);
    invr[r] = 1.0f / (nr + 1e-8f);
  }

  float* orow = out + (((size_t)b * NHW + h) * NHW) * NC;
  #pragma unroll
  for (int t = 0; t < 8; ++t) {
    int c = t * 16 + m;
    #pragma unroll
    for (int r = 0; r < 8; ++r) {
      int wp = wave * 16 + r + (hi << 3);
      orow[(size_t)wp * NC + c] = acc[t][r] * invr[r];
    }
  }
}

// ---------------------------------------------------------------------------
extern "C" void kernel_launch(void* const* d_in, const int* in_sizes, int n_in,
                              void* d_out, int out_size, void* d_ws, size_t ws_size,
                              hipStream_t stream) {
  const float* campos   = (const float*)d_in[0];
  const float* theta    = (const float*)d_in[1];
  const float* vertices = (const float*)d_in[2];
  const int*   faces    = (const int*)d_in[3];
  const float* memory   = (const float*)d_in[4];
  float* out = (float*)d_out;

  // workspace layout (floats): ~606 KB total
  float* wsf  = (float*)d_ws;
  float* ssum = wsf;                    // NV
  float* cnt  = ssum + NV;              // NV
  float* px   = cnt  + NV;              // NB*NV each
  float* py   = px  + NB * NV;
  float* s2d  = py  + NB * NV;
  float* zz   = s2d + NB * NV;
  float4* par4 = (float4*)(zz + NB * NV);        // NB*NV float4 (16B aligned)
  int*   sidx = (int*)(par4 + NB * NV);          // NB*NV ints
  _Float16* featT = (_Float16*)(sidx + NB * NV); // NB*NC*NV halfs (32B aligned)

  k_zero   <<<(NV + 255) / 256, 256, 0, stream>>>(ssum, cnt);
  k_edges  <<<(NF + 255) / 256, 256, 0, stream>>>(vertices, faces, ssum, cnt);
  k_project<<<(NB * NV + 255) / 256, 256, 0, stream>>>(campos, theta, vertices,
                                                       ssum, cnt, px, py, s2d, zz);
  k_sort   <<<NB, NV, 0, stream>>>(px, py, s2d, zz, par4, sidx);
  k_featT  <<<(NB * NC * NV + 255) / 256, 256, 0, stream>>>(memory, sidx, featT);
  dim3 grid(NHW, NB);
  k_main   <<<grid, 256, 0, stream>>>(par4, featT, out);
}
